// ResidueFiltration_76441827934276
// MI455X (gfx1250) — 8-bit matrix operand order repaired and checked by reading, compiled, not run on hardware
//
#include <hip/hip_runtime.h>
#include <hip/hip_bf16.h>
#include <stdint.h>

#define NPTS 4096
#define DIM  256
#define NEPS 4

typedef __attribute__((ext_vector_type(2))) float v2f;
typedef __attribute__((ext_vector_type(8))) float v8f;
typedef __attribute__((ext_vector_type(2))) int   v2i;
typedef __attribute__((ext_vector_type(4))) int   v4i;
typedef __attribute__((ext_vector_type(4))) float v4f;
typedef __attribute__((ext_vector_type(8))) int   v8i;

struct SelState {
    unsigned int prefix;
    unsigned int k;
    unsigned int nvalid;
    unsigned int thr_bits;
};

__global__ void rf_loss_sq(const float* __restrict__ residue,
                           const float* __restrict__ constraint,
                           float* __restrict__ out_losses,
                           float* __restrict__ sq) {
    int row  = (blockIdx.x * blockDim.x + threadIdx.x) >> 5;
    int lane = threadIdx.x & 31;
    const float* r = residue    + (size_t)row * DIM;
    const float* c = constraint + (size_t)row * DIM;
    float accL = 0.f, accS = 0.f;
#pragma unroll
    for (int k = 0; k < DIM; k += 32) {
        float rv = r[k + lane], cv = c[k + lane];
        float df = rv - cv;
        accL += df * df;
        accS += cv * cv;
    }
#pragma unroll
    for (int off = 16; off > 0; off >>= 1) {
        accL += __shfl_down(accL, off, 32);
        accS += __shfl_down(accS, off, 32);
    }
    if (lane == 0) {
        out_losses[row] = sqrtf(accL);
        sq[row]         = accS;
    }
}

__global__ void rf_vmask(const float* __restrict__ losses,
                         const float* __restrict__ eps,
                         float* __restrict__ out_vmask,
                         unsigned char* __restrict__ mask8) {
    int n = blockIdx.x * blockDim.x + threadIdx.x;
    int e = blockIdx.y;
    bool m = losses[n] <= eps[e];
    out_vmask[(size_t)e * NPTS + n] = m ? 1.f : 0.f;
    mask8[(size_t)e * NPTS + n]     = m ? 1 : 0;
}

__global__ void rf_dist_wmma(const float* __restrict__ X,
                             const float* __restrict__ sq,
                             float* __restrict__ dmat) {
    int wave = threadIdx.x >> 5;
    int lane = threadIdx.x & 31;
    int tile = blockIdx.x * 8 + wave;
    int ti = tile / (NPTS / 16);
    int tj = tile % (NPTS / 16);
    int half = lane >> 4;
    int mn   = lane & 15;

    const float* arow = X + (size_t)(ti * 16 + mn) * DIM + half * 2;
    const float* brow = X + (size_t)(tj * 16 + mn) * DIM + half * 2;

    v8f acc = {};
#pragma unroll 8
    for (int k = 0; k < DIM; k += 4) {
        v2f a = *(const v2f*)(arow + k);
        v2f b = *(const v2f*)(brow + k);
        acc = __builtin_amdgcn_wmma_f32_16x16x4_f32(
            false, a, false, b, (short)0, acc, false, false);
    }

    int col = tj * 16 + mn;
    float sqc = sq[col];
#pragma unroll
    for (int v = 0; v < 8; ++v) {
        int row = ti * 16 + half * 8 + v;
        float d2 = sq[row] + sqc - 2.f * acc[v];
        dmat[(size_t)row * NPTS + col] = sqrtf(fmaxf(d2, 0.f));
    }
}

__global__ void rf_init(unsigned int* __restrict__ hist, SelState* __restrict__ state) {
    int t = blockIdx.x * blockDim.x + threadIdx.x;
    if (t < NEPS * 2048) hist[t] = 0u;
    if (t < NEPS) state[t] = SelState{0u, 0u, 0u, 0u};
}

__global__ void rf_hist(const float* __restrict__ dmat,
                        const unsigned char* __restrict__ mask8,
                        unsigned int* __restrict__ hist,
                        const SelState* __restrict__ state,
                        int pass) {
    __shared__ unsigned int lh[2048];
    int e = blockIdx.y;
    for (int t = threadIdx.x; t < 2048; t += 256) lh[t] = 0u;
    __syncthreads();

    unsigned int prefix = state[e].prefix;
    const unsigned char* me = mask8 + (size_t)e * NPTS;
    unsigned int base = blockIdx.x * 256u + threadIdx.x;
    const unsigned int stride = 8192u * 256u;
#pragma unroll
    for (int it = 0; it < 8; ++it) {
        unsigned int idx = base + it * stride;
        unsigned int i = idx >> 12;
        unsigned int j = idx & 4095u;
        float v = dmat[idx];
        if (me[i] && me[j] && v > 0.f) {
            unsigned int u = __float_as_uint(v);
            unsigned int bin;
            bool match;
            if (pass == 0)      { bin = u >> 21;            match = true; }
            else if (pass == 1) { match = (u >> 21) == prefix; bin = (u >> 10) & 0x7FFu; }
            else                { match = (u >> 10) == prefix; bin = u & 0x3FFu; }
            if (match) atomicAdd(&lh[bin], 1u);
        }
    }
    __syncthreads();
    for (int t = threadIdx.x; t < 2048; t += 256) {
        unsigned int c = lh[t];
        if (c) atomicAdd(&hist[e * 2048 + t], c);
    }
}

__global__ void rf_scan(unsigned int* __restrict__ hist,
                        SelState* __restrict__ state,
                        int pass) {
    int e = threadIdx.x;
    if (e >= NEPS) return;
    unsigned int* h = hist + e * 2048;
    int nb = (pass == 2) ? 1024 : 2048;
    SelState s = state[e];
    if (pass == 0) {
        unsigned int total = 0;
        for (int b = 0; b < nb; ++b) total += h[b];
        s.nvalid = total;
        s.k = total ? (total - 1u) >> 1 : 0u;
    }
    if (s.nvalid == 0u) {
        s.thr_bits = 0x7F800000u;
    } else {
        unsigned int k = s.k, cum = 0u, sel = (unsigned int)(nb - 1);
        for (int b = 0; b < nb; ++b) {
            unsigned int c = h[b];
            if (cum + c > k) { sel = (unsigned int)b; break; }
            cum += c;
        }
        s.k = k - cum;
        if (pass == 0)      s.prefix = sel;
        else if (pass == 1) s.prefix = (s.prefix << 11) | sel;
        else                s.thr_bits = (s.prefix << 10) | sel;
    }
    for (int b = 0; b < 2048; ++b) h[b] = 0u;
    state[e] = s;
}

__global__ void rf_adj(const float* __restrict__ dmat,
                       const unsigned char* __restrict__ mask8,
                       const SelState* __restrict__ state,
                       float* __restrict__ outA,
                       unsigned char* __restrict__ A8) {
    int e = blockIdx.y;
    unsigned int idx4 = (blockIdx.x * 256u + threadIdx.x) * 4u;
    unsigned int i = idx4 >> 12;
    unsigned int j = idx4 & 4095u;
    float thr = __uint_as_float(state[e].thr_bits);
    const unsigned char* me = mask8 + (size_t)e * NPTS;
    bool mi = me[i] != 0;
    v4f dv = *(const v4f*)(dmat + idx4);
    v4f fa;
    unsigned int packed = 0u;
#pragma unroll
    for (int t = 0; t < 4; ++t) {
        bool a = mi && me[j + t] && (i != (j + t)) && (dv[t] <= thr);
        fa[t] = a ? 1.f : 0.f;
        packed |= (a ? 1u : 0u) << (8 * t);
    }
    size_t o = (size_t)e * NPTS * NPTS + idx4;
    *(v4f*)(outA + o) = fa;
    *(unsigned int*)(A8 + o) = packed;
}

__global__ void rf_tri_wmma(const unsigned char* __restrict__ A8all,
                            float* __restrict__ outTri) {
    int e = blockIdx.y;
    const unsigned char* A8 = A8all + (size_t)e * NPTS * NPTS;
    int wave = threadIdx.x >> 5;
    int lane = threadIdx.x & 31;
    int region = blockIdx.x * 8 + wave;
    const int RT = NPTS / 32;
    int ri = region / RT, rj = region % RT;
    int RB = ri * 32, CB = rj * 32;
    int half = lane >> 4;
    int mn   = lane & 15;
    int akb = half * 16;
    int bkb = half * 16;

    const unsigned char* ar0 = A8 + (size_t)(RB + mn)      * NPTS + akb;
    const unsigned char* ar1 = A8 + (size_t)(RB + 16 + mn) * NPTS + akb;
    const unsigned char* br0 = A8 + (size_t)(CB + mn)      * NPTS + bkb;
    const unsigned char* br1 = A8 + (size_t)(CB + 16 + mn) * NPTS + bkb;

    v8i acc00 = {}, acc01 = {}, acc10 = {}, acc11 = {};
    for (int k = 0; k < NPTS; k += 64) {
        if (k + 64 < NPTS) {
            __builtin_prefetch(ar0 + k + 64, 0, 0);
            __builtin_prefetch(ar1 + k + 64, 0, 0);
            __builtin_prefetch(br0 + k + 64, 0, 0);
            __builtin_prefetch(br1 + k + 64, 0, 0);
        }
        v2i a0p0 = *(const v2i*)(ar0 + k);
        v2i a0p1 = *(const v2i*)(ar0 + k + 8);
        v2i a0p2 = *(const v2i*)(ar0 + k + 32);
        v2i a0p3 = *(const v2i*)(ar0 + k + 40);
        v2i a1p0 = *(const v2i*)(ar1 + k);
        v2i a1p1 = *(const v2i*)(ar1 + k + 8);
        v2i a1p2 = *(const v2i*)(ar1 + k + 32);
        v2i a1p3 = *(const v2i*)(ar1 + k + 40);
        v8i a0 = {a0p0[0], a0p0[1], a0p1[0], a0p1[1],
                  a0p2[0], a0p2[1], a0p3[0], a0p3[1]};
        v8i a1 = {a1p0[0], a1p0[1], a1p1[0], a1p1[1],
                  a1p2[0], a1p2[1], a1p3[0], a1p3[1]};
        v4i b0lo = *(const v4i*)(br0 + k);
        v4i b0hi = *(const v4i*)(br0 + k + 32);
        v4i b1lo = *(const v4i*)(br1 + k);
        v4i b1hi = *(const v4i*)(br1 + k + 32);
        v8i b0 = {b0lo[0], b0lo[1], b0lo[2], b0lo[3],
                  b0hi[0], b0hi[1], b0hi[2], b0hi[3]};
        v8i b1 = {b1lo[0], b1lo[1], b1lo[2], b1lo[3],
                  b1hi[0], b1hi[1], b1hi[2], b1hi[3]};

        acc00 = __builtin_amdgcn_wmma_i32_16x16x64_iu8(false, a0, false, b0, acc00, false, false);
        acc01 = __builtin_amdgcn_wmma_i32_16x16x64_iu8(false, a0, false, b1, acc01, false, false);
        acc10 = __builtin_amdgcn_wmma_i32_16x16x64_iu8(false, a1, false, b0, acc10, false, false);
        acc11 = __builtin_amdgcn_wmma_i32_16x16x64_iu8(false, a1, false, b1, acc11, false, false);
    }

    float* tri = outTri + (size_t)e * NPTS * NPTS;
#pragma unroll
    for (int v = 0; v < 8; ++v) {
        int r0 = RB + half * 8 + v;
        int r1 = r0 + 16;
        int c0 = CB + mn;
        int c1 = c0 + 16;
        { size_t o = (size_t)r0 * NPTS + c0; tri[o] = A8[o] ? (float)acc00[v] : 0.f; }
        { size_t o = (size_t)r0 * NPTS + c1; tri[o] = A8[o] ? (float)acc01[v] : 0.f; }
        { size_t o = (size_t)r1 * NPTS + c0; tri[o] = A8[o] ? (float)acc10[v] : 0.f; }
        { size_t o = (size_t)r1 * NPTS + c1; tri[o] = A8[o] ? (float)acc11[v] : 0.f; }
    }
}

extern "C" void kernel_launch(void* const* d_in, const int* in_sizes, int n_in,
                              void* d_out, int out_size, void* d_ws, size_t ws_size,
                              hipStream_t stream) {
    const float* residue    = (const float*)d_in[0];
    const float* constraint = (const float*)d_in[1];
    const float* eps        = (const float*)d_in[2];

    float* out        = (float*)d_out;
    float* out_losses = out;
    float* out_vmask  = out + NPTS;
    float* out_A      = out_vmask + (size_t)NEPS * NPTS;
    float* out_tri    = out_A + (size_t)NEPS * NPTS * NPTS;

    char* ws = (char*)d_ws;
    float*         dmat  = (float*)ws;
    unsigned char* A8    = (unsigned char*)(ws + (size_t)NPTS * NPTS * 4);
    unsigned int*  hist  = (unsigned int*)(ws + (size_t)NPTS * NPTS * 4
                                               + (size_t)NEPS * NPTS * NPTS);
    SelState*      state = (SelState*)((char*)hist + NEPS * 2048 * 4);
    float*         sq    = (float*)((char*)state + NEPS * sizeof(SelState));
    unsigned char* mask8 = (unsigned char*)(sq + NPTS);

    rf_loss_sq<<<NPTS * 32 / 256, 256, 0, stream>>>(residue, constraint, out_losses, sq);
    rf_vmask<<<dim3(NPTS / 256, NEPS), 256, 0, stream>>>(out_losses, eps, out_vmask, mask8);
    rf_dist_wmma<<<(NPTS / 16) * (NPTS / 16) / 8, 256, 0, stream>>>(constraint, sq, dmat);
    rf_init<<<NEPS * 2048 / 256, 256, 0, stream>>>(hist, state);
    for (int pass = 0; pass < 3; ++pass) {
        rf_hist<<<dim3(8192, NEPS), 256, 0, stream>>>(dmat, mask8, hist, state, pass);
        rf_scan<<<1, 32, 0, stream>>>(hist, state, pass);
    }
    rf_adj<<<dim3(NPTS * NPTS / 4 / 256, NEPS), 256, 0, stream>>>(dmat, mask8, state, out_A, A8);
    rf_tri_wmma<<<dim3((NPTS / 32) * (NPTS / 32) / 8, NEPS), 256, 0, stream>>>(A8, out_tri);
}
